// RopeAttention_180388626700
// MI455X (gfx1250) — matrix operand order repaired and checked by reading, not compiled, not run on hardware
//
#include <hip/hip_runtime.h>

#define B_ 2
#define S_ 2048
#define H_ 2048
#define NH_ 16
#define HD_ 128

typedef _Float16 v16h __attribute__((ext_vector_type(16)));
typedef _Float16 half8 __attribute__((ext_vector_type(8)));
typedef float v8f __attribute__((ext_vector_type(8)));

union F16x16 { v16h v; half8 h[2]; };

__device__ __forceinline__ v8f wmma16(v16h a, v16h b, v8f c) {
  // D = A(16x32 f16) x B(32x16 f16) + C(16x16 f32)
  return __builtin_amdgcn_wmma_f32_16x16x32_f16(false, a, false, b, (short)0, c,
                                                false, false);
}

// A-matrix fragment (16x32 f16): lane L holds row L%16; halves j=0..15 -> k=(L/16)*16+j.
__device__ __forceinline__ v16h fragA(const _Float16* base, int strideH, int lane) {
  const _Float16* p = base + (size_t)(lane & 15) * strideH + ((lane >> 4) << 4);
  F16x16 f;
  f.h[0] = *(const half8*)(p);
  f.h[1] = *(const half8*)(p + 8);
  return f.v;
}

// B-matrix fragment (32x16 f16): lane L holds column n=L%16; halves j=0..15 -> k=(L/16)*16+j.
// "base" points at column 0 stored as a contiguous row of K values.
__device__ __forceinline__ v16h fragB(const _Float16* base, int strideH, int lane) {
  const _Float16* p = base + (size_t)(lane & 15) * strideH + ((lane >> 4) << 4);
  F16x16 f;
  f.h[0] = *(const half8*)(p);
  f.h[1] = *(const half8*)(p + 8);
  return f.v;
}

__device__ __forceinline__ half8 cvt8(float4 a, float4 b) {
  half8 h;
  h[0] = (_Float16)a.x; h[1] = (_Float16)a.y; h[2] = (_Float16)a.z; h[3] = (_Float16)a.w;
  h[4] = (_Float16)b.x; h[5] = (_Float16)b.y; h[6] = (_Float16)b.z; h[7] = (_Float16)b.w;
  return h;
}
__device__ __forceinline__ void ld2f4(float4* r, const float* p) {
  const float4* q = (const float4*)p;
  r[0] = q[0]; r[1] = q[1];
}
__device__ __forceinline__ void commit8(_Float16* dst, const float4* r) {
  *(half8*)dst = cvt8(r[0], r[1]);
}

// A-tile segment issue: f32 path stages 8 floats in VGPRs (converted at commit);
// f16 path copies 16B global->LDS directly via the CDNA5 async-to-LDS DMA path
// (tracked by ASYNCcnt; completed with s_wait_asynccnt before the tile barrier).
__device__ __forceinline__ void issueA(float4* r, const float* src, _Float16*) {
  ld2f4(r, src);
}
__device__ __forceinline__ void issueA(float4* r, const _Float16* src, _Float16* ldsDst) {
  (void)r;
  unsigned lds = (unsigned)(unsigned long long)ldsDst;  // low 32 bits = LDS byte offset
  asm volatile("global_load_async_to_lds_b128 %0, %1, off" ::"v"(lds), "v"(src)
               : "memory");
}

__device__ __forceinline__ void stOut(float* p, float v) { *p = v; }
__device__ __forceinline__ void stOut(_Float16* p, float v) { *p = (_Float16)v; }

// ---------------------------------------------------------------------------
// C[M,N] = A[M,K] @ W[N,K]^T (torch Linear), f32 accumulate via WMMA.
// 128x128 block tile, BK=32, 8 waves (2x4), each wave 64x32 via 8 WMMA accs.
// Double-buffered LDS, one barrier per K-step: issue next-tile loads, compute
// current tile, then commit fills (cvt+ds_store for f32 / asynccnt wait for f16).
// ---------------------------------------------------------------------------
template <typename AT, typename OT>
__global__ __launch_bounds__(256) void gemm_xWT(const AT* __restrict__ A,
                                                const float* __restrict__ W,
                                                OT* __restrict__ C,
                                                int M, int N, int K) {
  constexpr bool ASYNC_A = (sizeof(AT) == 2);
  __shared__ __align__(16) _Float16 As[2][128 * 40];
  __shared__ __align__(16) _Float16 Bs[2][128 * 40];
  const int t = threadIdx.x;
  const int w = t >> 5, lane = t & 31;
  const int wm = w >> 2, wn = w & 3;  // 2x4 wave grid
  const int m0 = blockIdx.y * 128, n0 = blockIdx.x * 128;

  // Two 8-element segments per thread: rows r and r+64, same k-chunk.
  const int row0 = t >> 2, kc0 = (t & 3) << 3;
  const AT* aSeg0 = A + (size_t)(m0 + row0) * K + kc0;
  const AT* aSeg1 = A + (size_t)(m0 + row0 + 64) * K + kc0;
  const float* wSeg0 = W + (size_t)(n0 + row0) * K + kc0;
  const float* wSeg1 = W + (size_t)(n0 + row0 + 64) * K + kc0;
  const int ldsOff0 = row0 * 40 + kc0;
  const int ldsOff1 = (row0 + 64) * 40 + kc0;

  float4 ar[2][2], wr[2][2];

  // Prologue: fill tile 0.
  issueA(ar[0], aSeg0, &As[0][ldsOff0]);
  issueA(ar[1], aSeg1, &As[0][ldsOff1]);
  ld2f4(wr[0], wSeg0);
  ld2f4(wr[1], wSeg1);
  if (!ASYNC_A) {
    commit8(&As[0][ldsOff0], ar[0]);
    commit8(&As[0][ldsOff1], ar[1]);
  }
  commit8(&Bs[0][ldsOff0], wr[0]);
  commit8(&Bs[0][ldsOff1], wr[1]);
  if (ASYNC_A) asm volatile("s_wait_asynccnt 0x0" ::: "memory");
  __syncthreads();

  v8f acc[4][2] = {};
  for (int k0 = 0; k0 < K; k0 += 32) {
    const int cur = (k0 >> 5) & 1, nxt = cur ^ 1;
    const bool has = (k0 + 32) < K;
    if (has) {  // issue next tile's global traffic before computing
      issueA(ar[0], aSeg0 + (k0 + 32), &As[nxt][ldsOff0]);
      issueA(ar[1], aSeg1 + (k0 + 32), &As[nxt][ldsOff1]);
      ld2f4(wr[0], wSeg0 + (k0 + 32));
      ld2f4(wr[1], wSeg1 + (k0 + 32));
    }
    const _Float16* aB = As[cur] + (wm * 64) * 40;
    const _Float16* bB = Bs[cur] + (wn * 32) * 40;
    v16h bf0 = fragB(bB, 40, lane);
    v16h bf1 = fragB(bB + 16 * 40, 40, lane);
#pragma unroll
    for (int mi = 0; mi < 4; ++mi) {
      v16h af = fragA(aB + mi * 16 * 40, 40, lane);
      acc[mi][0] = wmma16(af, bf0, acc[mi][0]);
      acc[mi][1] = wmma16(af, bf1, acc[mi][1]);
    }
    if (has) {
      if (!ASYNC_A) {
        commit8(&As[nxt][ldsOff0], ar[0]);
        commit8(&As[nxt][ldsOff1], ar[1]);
      }
      commit8(&Bs[nxt][ldsOff0], wr[0]);
      commit8(&Bs[nxt][ldsOff1], wr[1]);
      if (ASYNC_A) asm volatile("s_wait_asynccnt 0x0" ::: "memory");
    }
    __syncthreads();
  }

  const int g = lane >> 4, ln = lane & 15;
#pragma unroll
  for (int mi = 0; mi < 4; ++mi)
#pragma unroll
    for (int ni = 0; ni < 2; ++ni)
#pragma unroll
      for (int i = 0; i < 8; ++i) {
        int row = m0 + wm * 64 + mi * 16 + i + 8 * g;  // C: lane holds rows i+8g
        int col = n0 + wn * 32 + ni * 16 + ln;         //    column = lane%16
        stOut(&C[(size_t)row * N + col], acc[mi][ni][i]);
      }
}

// ---------------------------------------------------------------------------
// RoPE on Q (in place) and on V (faithful to reference), V written transposed
// to [B, NH, HD, S] so PV WMMA B-fragments are contiguous along kv.
// ---------------------------------------------------------------------------
__global__ __launch_bounds__(256) void rope_qv(_Float16* __restrict__ q,
                                               const _Float16* __restrict__ v,
                                               _Float16* __restrict__ vT) {
  int idx = blockIdx.x * 256 + threadIdx.x;  // over B*S*NH*64
  int j = idx & 63;
  int hh = (idx >> 6) & (NH_ - 1);
  int s = (idx >> 10) & (S_ - 1);
  int b = idx >> 21;
  float theta = (float)s * __powf(10000.0f, -(float)j * (1.0f / 64.0f));
  float sn, cs;
  __sincosf(theta, &sn, &cs);
  size_t base = ((size_t)(b * S_ + s) * NH_ + hh) * HD_;
  float q0 = (float)q[base + j], q1 = (float)q[base + j + 64];
  q[base + j]      = (_Float16)(q0 * cs - q1 * sn);
  q[base + j + 64] = (_Float16)(q1 * cs + q0 * sn);
  float v0 = (float)v[base + j], v1 = (float)v[base + j + 64];
  size_t tb = (size_t)(b * NH_ + hh) * HD_;
  vT[(tb + j) * S_ + s]      = (_Float16)(v0 * cs - v1 * sn);
  vT[(tb + j + 64) * S_ + s] = (_Float16)(v1 * cs + v0 * sn);
}

// ---------------------------------------------------------------------------
// Flash attention: block = 8 fully independent waves, each owning 16 q rows.
// Per 32 kv columns: 8 WMMAs for Q.K^T, f32 online softmax with multiplicative
// mask, 8 WMMAs for P.V. P relayout (C->A frag) uses wave-private LDS; LDS ops
// from one wave execute in order (DScnt), so no workgroup barrier is needed.
// ---------------------------------------------------------------------------
__global__ __launch_bounds__(256) void attn_fa(const _Float16* __restrict__ q,
                                               const _Float16* __restrict__ k,
                                               const _Float16* __restrict__ vT,
                                               const float* __restrict__ mask,
                                               _Float16* __restrict__ out) {
  __shared__ __align__(16) _Float16 pbuf[8][16 * 40];  // per-wave C->A relayout
  const int t = threadIdx.x, w = t >> 5, lane = t & 31;
  const int g = lane >> 4, ln = lane & 15;
  const int qt = blockIdx.x, hh = blockIdx.y, b = blockIdx.z;
  const int qBase = qt * 128 + w * 16;
  const float scaling = 0.08838834764831845f;  // 1/sqrt(128)

  v16h qf[4];
  {
    const _Float16* qrow = q + (size_t)(b * S_ + qBase + ln) * H_ + hh * HD_;
#pragma unroll
    for (int ks = 0; ks < 4; ++ks) {  // A-fragment pattern
      const _Float16* p = qrow + ks * 32 + g * 16;
      F16x16 f; f.h[0] = *(const half8*)p; f.h[1] = *(const half8*)(p + 8);
      qf[ks] = f.v;
    }
  }
  v8f o[8] = {};
  float mrow[8], lrow[8];
#pragma unroll
  for (int i = 0; i < 8; ++i) { mrow[i] = -1e30f; lrow[i] = 0.0f; }
  const float* maskb = mask + (size_t)b * S_ * S_;
  const _Float16* krow0 = k + (size_t)(b * S_) * H_ + hh * HD_;
  const _Float16* vbase = vT + (size_t)(b * NH_ + hh) * HD_ * S_;

  for (int kv0 = 0; kv0 < S_; kv0 += 32) {
    v8f st[2] = {};
#pragma unroll
    for (int tt = 0; tt < 2; ++tt)
#pragma unroll
      for (int ks = 0; ks < 4; ++ks) {  // B-frag: column n = kv row, contiguous k
        const _Float16* p = krow0 + (size_t)(kv0 + tt * 16 + ln) * H_ + ks * 32 + g * 16;
        F16x16 f; f.h[0] = *(const half8*)p; f.h[1] = *(const half8*)(p + 8);
        st[tt] = wmma16(qf[ks], f.v, st[tt]);
      }
    // scale + multiplicative mask (faithful), then online softmax in f32
    float ps[2][8];
#pragma unroll
    for (int tt = 0; tt < 2; ++tt)
#pragma unroll
      for (int i = 0; i < 8; ++i) {
        int row = qBase + i + 8 * g;
        int col = kv0 + tt * 16 + ln;
        ps[tt][i] = st[tt][i] * scaling * maskb[(size_t)row * S_ + col];
      }
    float alpha[8];
#pragma unroll
    for (int i = 0; i < 8; ++i) {
      float vmax = fmaxf(ps[0][i], ps[1][i]);
#pragma unroll
      for (int off = 1; off < 16; off <<= 1)
        vmax = fmaxf(vmax, __shfl_xor(vmax, off, 16));
      float nm = fmaxf(mrow[i], vmax);
      alpha[i] = __expf(mrow[i] - nm);
      mrow[i] = nm;
      float p0 = __expf(ps[0][i] - nm);
      float p1 = __expf(ps[1][i] - nm);
      ps[0][i] = p0; ps[1][i] = p1;
      float rs = p0 + p1;
#pragma unroll
      for (int off = 1; off < 16; off <<= 1)
        rs += __shfl_xor(rs, off, 16);
      lrow[i] = lrow[i] * alpha[i] + rs;
    }
#pragma unroll
    for (int ni = 0; ni < 8; ++ni)
#pragma unroll
      for (int i = 0; i < 8; ++i) o[ni][i] *= alpha[i];
    // C-layout P -> A-fragment via wave-private LDS (same-wave DS ops in order)
#pragma unroll
    for (int tt = 0; tt < 2; ++tt)
#pragma unroll
      for (int i = 0; i < 8; ++i)
        pbuf[w][(i + 8 * g) * 40 + tt * 16 + ln] = (_Float16)ps[tt][i];
    asm volatile("" ::: "memory");  // keep store->load program order
    v16h pf = fragA(&pbuf[w][0], 40, lane);
#pragma unroll
    for (int ni = 0; ni < 8; ++ni) {  // B-frag: column n = head dim, contiguous kv
      const _Float16* p = vbase + (size_t)(ni * 16 + ln) * S_ + kv0 + g * 16;
      F16x16 f; f.h[0] = *(const half8*)p; f.h[1] = *(const half8*)(p + 8);
      o[ni] = wmma16(pf, f.v, o[ni]);
    }
  }
#pragma unroll
  for (int i = 0; i < 8; ++i) lrow[i] = 1.0f / lrow[i];
#pragma unroll
  for (int ni = 0; ni < 8; ++ni)
#pragma unroll
    for (int i = 0; i < 8; ++i) {
      int row = qBase + i + 8 * g;
      int col = hh * HD_ + ni * 16 + ln;
      out[(size_t)(b * S_ + row) * H_ + col] = (_Float16)(o[ni][i] * lrow[i]);
    }
}

// ---------------------------------------------------------------------------
extern "C" void kernel_launch(void* const* d_in, const int* in_sizes, int n_in,
                              void* d_out, int out_size, void* d_ws, size_t ws_size,
                              hipStream_t stream) {
  const float* x    = (const float*)d_in[0];
  const float* mask = (const float*)d_in[1];
  const float* Wq   = (const float*)d_in[2];
  const float* Wk   = (const float*)d_in[3];
  const float* Wv   = (const float*)d_in[4];
  const float* Wo   = (const float*)d_in[5];
  float* outp = (float*)d_out;

  const size_t MN = (size_t)B_ * S_ * H_;  // 8,388,608 elements
  _Float16* q16   = (_Float16*)d_ws;       // 5 f16 buffers = 80 MB total
  _Float16* k16   = q16 + MN;
  _Float16* v16   = k16 + MN;
  _Float16* vT16  = v16 + MN;
  _Float16* att16 = vT16 + MN;

  dim3 blk(256);
  dim3 gGemm(H_ / 128, (B_ * S_) / 128);  // (16, 32)

  gemm_xWT<float, _Float16><<<gGemm, blk, 0, stream>>>(x, Wq, q16, B_ * S_, H_, H_);
  gemm_xWT<float, _Float16><<<gGemm, blk, 0, stream>>>(x, Wk, k16, B_ * S_, H_, H_);
  gemm_xWT<float, _Float16><<<gGemm, blk, 0, stream>>>(x, Wv, v16, B_ * S_, H_, H_);

  int ropeN = B_ * S_ * NH_ * 64;  // one thread per rotated pair
  rope_qv<<<ropeN / 256, blk, 0, stream>>>(q16, v16, vT16);

  dim3 gAttn(S_ / 128, NH_, B_);
  attn_fa<<<gAttn, blk, 0, stream>>>(q16, k16, vT16, mask, att16);

  gemm_xWT<_Float16, float><<<gGemm, blk, 0, stream>>>(att16, Wo, outp, B_ * S_, H_, H_);
}
